// GATClassifier_5076651344319
// MI455X (gfx1250) — compile-verified, matrix operand order checked by reading, not run on hardware
//
#include <hip/hip_runtime.h>
#include <math.h>
#include <stdint.h>

// ---------------------------------------------------------------------------
// Types for CDNA5 WMMA (wave32, 16x16x32 bf16 -> f32)
// ---------------------------------------------------------------------------
typedef __bf16 bf16_t;
typedef bf16_t v16bf __attribute__((ext_vector_type(16)));
typedef bf16_t v8bf  __attribute__((ext_vector_type(8)));
typedef float  v8f   __attribute__((ext_vector_type(8)));
typedef int    v4i32 __attribute__((vector_size(16)));  // GCC vector: matches builtin's V4i

#define BM 128
#define BN 64
#define BK 32
#define AS_STRIDE 36  // fp32 elems per LDS A row: 144B, 16B-aligned, bank-skewed

#define GLOBAL_AS __attribute__((address_space(1)))
#define LDS_AS    __attribute__((address_space(3)))

#if __has_builtin(__builtin_amdgcn_global_load_async_to_lds_b128) && \
    __has_builtin(__builtin_amdgcn_s_wait_asynccnt)
#define USE_ASYNC 1
#else
#define USE_ASYNC 0
#endif

__device__ __forceinline__ GLOBAL_AS v4i32* to_global_v4(const void* p) {
  return (GLOBAL_AS v4i32*)(unsigned long long)(uintptr_t)p;
}
__device__ __forceinline__ LDS_AS v4i32* to_lds_v4(void* p) {
  // generic LDS pointers carry the LDS offset in the low 32 bits
  return (LDS_AS v4i32*)(unsigned)(uintptr_t)p;
}

// ---------------------------------------------------------------------------
// Weight pre-pack: fp32 [K][N] row-major -> bf16 fragment-major.
// Flat index F = ((kt*NT + nt)*32 + lane)*16 + j, where per ISA B layout:
//   i = j&7, hi = j>>3, half = lane>>4, lr = lane&15
//   k = kt*32 + half*8 + hi*16 + i ; n = nt*16 + lr
// So each lane's v16bf fragment is 32 contiguous bytes -> 2x global_load_b128.
// ---------------------------------------------------------------------------
__global__ __launch_bounds__(256) void pack_weight_kernel(
    const float* __restrict__ W, bf16_t* __restrict__ Wf, int K, int N) {
  long long F = (long long)blockIdx.x * blockDim.x + threadIdx.x;
  if (F >= (long long)K * N) return;
  int j    = (int)(F & 15);
  int lane = (int)((F >> 4) & 31);
  long long tile = F >> 9;
  int NT = N >> 4;
  int nt = (int)(tile % NT);
  int kt = (int)(tile / NT);
  int half = lane >> 4, lr = lane & 15;
  int i = j & 7, hi = j >> 3;
  int k = (kt << 5) + (half << 3) + (hi << 4) + i;
  int n = (nt << 4) + lr;
  Wf[F] = (bf16_t)W[(size_t)k * N + n];
}

__device__ __forceinline__ v16bf load_bfrag(const bf16_t* __restrict__ Bf,
                                            int ktile, int ntile, int NT,
                                            int lane) {
  const v8bf* p =
      (const v8bf*)(Bf + (((size_t)ktile * NT + ntile) * 32 + lane) * 16);
  v8bf lo = p[0];
  v8bf hi = p[1];
  return __builtin_shufflevector(lo, hi, 0, 1, 2, 3, 4, 5, 6, 7, 8, 9, 10, 11,
                                 12, 13, 14, 15);
}

// ---------------------------------------------------------------------------
// WMMA GEMM: C[M,N] = A[M,K] @ B[K,N] (+bias)(+relu).
// A fp32 in global, async-staged (double-buffered) into LDS as fp32;
// B pre-packed bf16 fragment-major in global (L2-resident weights).
// K % 32 == 0, N % 64 == 0 (true for all 4 GEMMs); M guarded at the store.
// Out-of-range rows are CLAMPED during staging (uniform control flow, no
// exec-mask diamonds around the async loads); their results are discarded.
// 256 threads = 8 waves; each wave owns a 32x32 sub-tile (2x2 WMMA frags).
// EPI: 0 = none, 1 = +bias, 2 = +bias then relu
// ---------------------------------------------------------------------------
template <int EPI>
__global__ __launch_bounds__(256) void wmma_gemm_kernel(
    const float* __restrict__ A, const bf16_t* __restrict__ Bf,
    const float* __restrict__ bias, float* __restrict__ C,
    int M, int N, int K) {
  __shared__ float As[2][BM][AS_STRIDE];

  const int tid  = threadIdx.x;
  const int lane = tid & 31;
  const int wid  = tid >> 5;
  const int wm   = (wid & 3) << 5;  // wave row offset (0,32,64,96)
  const int wn   = (wid >> 2) << 5; // wave col offset (0,32)
  const int half = lane >> 4;       // 0/1 (K-halves per ISA layout)
  const int lr   = lane & 15;

  const int bm = blockIdx.x * BM;
  const int bn = blockIdx.y * BN;
  const int NT = N >> 4;
  const int ntile0 = (bn + wn) >> 4;

  v8f acc[2][2];
  acc[0][0] = {}; acc[0][1] = {}; acc[1][0] = {}; acc[1][1] = {};

  // Per-thread staged rows (clamped once; uniform inner control flow).
  int srow[4];
#pragma unroll
  for (int i = 0; i < 4; ++i) {
    int gm = bm + ((tid + (i << 8)) >> 3);
    srow[i] = gm < M ? gm : M - 1;
  }

  // Stage one 128x32 fp32 A tile into LDS buffer `buf`.
  // 1024 16B-chunks, 4 per thread, no branches.
  auto stage = [&](int k0, int buf) {
#pragma unroll
    for (int i = 0; i < 4; ++i) {
      int j = tid + (i << 8);
      int r = j >> 3;            // row in tile
      int c = (j & 7) << 2;      // fp32 column (16B chunk)
#if USE_ASYNC
      __builtin_amdgcn_global_load_async_to_lds_b128(
          to_global_v4(A + (size_t)srow[i] * K + k0 + c),
          to_lds_v4(&As[buf][r][c]), 0, 0);
#else
      *(float4*)&As[buf][r][c] =
          *(const float4*)(A + (size_t)srow[i] * K + k0 + c);
#endif
    }
  };

  const int nk = K >> 5;
  stage(0, 0);

  for (int kt = 0; kt < nk; ++kt) {
    const int cur = kt & 1;
    if (kt + 1 < nk) {
      stage((kt + 1) << 5, cur ^ 1);
#if USE_ASYNC
      __builtin_amdgcn_s_wait_asynccnt(4);  // tile kt landed; kt+1 in flight
#endif
      // keep next B fragments warm in L2
      __builtin_prefetch(Bf + (((size_t)(kt + 1) * NT + ntile0) * 32 + lane) * 16,
                         0, 1);
    } else {
#if USE_ASYNC
      __builtin_amdgcn_s_wait_asynccnt(0);
#endif
    }
    __syncthreads();

    // B fragments straight from global (2x 16B per fragment, coalesced).
    v16bf bfrag[2];
#pragma unroll
    for (int ni = 0; ni < 2; ++ni)
      bfrag[ni] = load_bfrag(Bf, kt, ntile0 + ni, NT, lane);

    // A fragments from LDS (wide ds_load_b128s) + register fp32->bf16 convert.
    v16bf afrag[2];
#pragma unroll
    for (int t = 0; t < 2; ++t) {
      const float* ar = &As[cur][wm + (t << 4) + lr][half << 3];
      v16bf a;
#pragma unroll
      for (int i = 0; i < 8; ++i) {
        a[i]     = (bf16_t)ar[i];
        a[i + 8] = (bf16_t)ar[16 + i];
      }
      afrag[t] = a;
    }

#pragma unroll
    for (int mi = 0; mi < 2; ++mi)
#pragma unroll
      for (int ni = 0; ni < 2; ++ni)
        acc[mi][ni] = __builtin_amdgcn_wmma_f32_16x16x32_bf16(
            false, afrag[mi], false, bfrag[ni], (short)0, acc[mi][ni],
            false, false);

    __syncthreads();
  }

  // Epilogue + store (C/D layout: lanes 0-15 -> M=r, lanes 16-31 -> M=r+8).
  const int mbase = bm + wm + (half << 3);
  if (bm + BM <= M) {
    // Fast path: full tile, no guards (390 of 391 blocks).
#pragma unroll
    for (int mi = 0; mi < 2; ++mi) {
#pragma unroll
      for (int ni = 0; ni < 2; ++ni) {
        int gn = bn + wn + (ni << 4) + lr;
        float bv = (EPI >= 1) ? bias[gn] : 0.f;
#pragma unroll
        for (int r = 0; r < 8; ++r) {
          float v = acc[mi][ni][r] + bv;
          if (EPI == 2) v = fmaxf(v, 0.f);
          C[(size_t)(mbase + (mi << 4) + r) * N + gn] = v;
        }
      }
    }
  } else {
#pragma unroll
    for (int mi = 0; mi < 2; ++mi) {
#pragma unroll
      for (int ni = 0; ni < 2; ++ni) {
        int gn = bn + wn + (ni << 4) + lr;
        float bv = (EPI >= 1) ? bias[gn] : 0.f;
#pragma unroll
        for (int r = 0; r < 8; ++r) {
          int gm = mbase + (mi << 4) + r;
          if (gm < M) {
            float v = acc[mi][ni][r] + bv;
            if (EPI == 2) v = fmaxf(v, 0.f);
            C[(size_t)gm * N + gn] = v;
          }
        }
      }
    }
  }
}

// ---------------------------------------------------------------------------
// Attention coefficients: als[n,h] = <xw[n,h,:], a_s[h,:]>, same for ald.
// One wave per (node, head); shuffle reduction.
// ---------------------------------------------------------------------------
__global__ __launch_bounds__(256) void attn_coef_kernel(
    const float* __restrict__ xw, const float* __restrict__ a_s,
    const float* __restrict__ a_d, float* __restrict__ als,
    float* __restrict__ ald, int n_nodes, int H, int C) {
  long long w = ((long long)blockIdx.x * blockDim.x + threadIdx.x) >> 5;
  int lane = threadIdx.x & 31;
  if (w >= (long long)n_nodes * H) return;
  int node = (int)(w / H);
  int h    = (int)(w % H);
  const float* row = xw + (size_t)node * H * C + (size_t)h * C;
  float s = 0.f, d = 0.f;
  for (int c = lane; c < C; c += 32) {
    float v = row[c];
    s += v * a_s[h * C + c];
    d += v * a_d[h * C + c];
  }
#pragma unroll
  for (int off = 16; off; off >>= 1) {
    s += __shfl_xor(s, off);
    d += __shfl_xor(d, off);
  }
  if (lane == 0) { als[w] = s; ald[w] = d; }
}

// ---------------------------------------------------------------------------
// Monotone float <-> uint32 encoding for atomic float max via atomicMax(u32).
// ---------------------------------------------------------------------------
__device__ __forceinline__ unsigned f32_ord(float f) {
  unsigned u = __float_as_uint(f);
  return (u & 0x80000000u) ? ~u : (u | 0x80000000u);
}
__device__ __forceinline__ float ord_f32(unsigned u) {
  u = (u & 0x80000000u) ? (u & 0x7FFFFFFFu) : ~u;
  return __uint_as_float(u);
}
#define ORD_NEG_INF 0x007FFFFFu  // f32_ord(-inf)

__device__ __forceinline__ void edge_nodes(const long long* __restrict__ ei,
                                           long long E, long long e,
                                           int& src, int& dst) {
  if (e < E) { src = (int)ei[e]; dst = (int)ei[E + e]; }
  else       { src = dst = (int)(e - E); }  // appended self-loops
}

__device__ __forceinline__ float leaky02(float v) {
  return v > 0.f ? v : 0.2f * v;
}

// Pass 1: segment max over dst
__global__ __launch_bounds__(256) void edge_max_kernel(
    const long long* __restrict__ ei, int E, int Nn, int H,
    const float* __restrict__ als, const float* __restrict__ ald,
    unsigned* __restrict__ mx) {
  long long e = (long long)blockIdx.x * blockDim.x + threadIdx.x;
  long long Et = (long long)E + Nn;
  if (e >= Et) return;
  int s, d; edge_nodes(ei, E, e, s, d);
  for (int h = 0; h < H; ++h) {
    float v = leaky02(als[(size_t)s * H + h] + ald[(size_t)d * H + h]);
    atomicMax(&mx[(size_t)d * H + h], f32_ord(v));
  }
}

// Pass 2: ex = exp(e - max[dst]); denom[dst] += ex; store ex
__global__ __launch_bounds__(256) void edge_expsum_kernel(
    const long long* __restrict__ ei, int E, int Nn, int H,
    const float* __restrict__ als, const float* __restrict__ ald,
    const unsigned* __restrict__ mx, float* __restrict__ ex,
    float* __restrict__ dn) {
  long long e = (long long)blockIdx.x * blockDim.x + threadIdx.x;
  long long Et = (long long)E + Nn;
  if (e >= Et) return;
  int s, d; edge_nodes(ei, E, e, s, d);
  for (int h = 0; h < H; ++h) {
    float v = leaky02(als[(size_t)s * H + h] + ald[(size_t)d * H + h]);
    float m = ord_f32(mx[(size_t)d * H + h]);
    float x = expf(v - m);
    ex[(size_t)e * H + h] = x;
    atomicAdd(&dn[(size_t)d * H + h], x);
  }
}

// Pass 3: agg[dst, h, c] += alpha[e,h] * xw[src, h, c]. One wave per edge.
__global__ __launch_bounds__(256) void edge_agg_kernel(
    const long long* __restrict__ ei, int E, int Nn, int H, int C,
    const float* __restrict__ xw, const float* __restrict__ ex,
    const float* __restrict__ dn, float* __restrict__ agg) {
  long long w = (long long)blockIdx.x * (blockDim.x >> 5) + (threadIdx.x >> 5);
  int lane = threadIdx.x & 31;
  long long Et = (long long)E + Nn;
  if (w >= Et) return;
  int s, d; edge_nodes(ei, E, w, s, d);
  float alpha[4];
  for (int h = 0; h < H; ++h)
    alpha[h] = ex[(size_t)w * H + h] / dn[(size_t)d * H + h];
  int HC = H * C;
  const float* srow = xw + (size_t)s * HC;
  float* drow = agg + (size_t)d * HC;
  for (int i = lane; i < HC; i += 32) {
    int h = i / C;
    atomicAdd(&drow[i], srow[i] * alpha[h]);
  }
}

// ---------------------------------------------------------------------------
// Block reductions + LayerNorm variants
// ---------------------------------------------------------------------------
__device__ __forceinline__ float block_sum(float v, float* red, int nwaves) {
  int lane = threadIdx.x & 31, wid = threadIdx.x >> 5;
#pragma unroll
  for (int off = 16; off; off >>= 1) v += __shfl_xor(v, off);
  if (lane == 0) red[wid] = v;
  __syncthreads();
  float s = 0.f;
  for (int i = 0; i < nwaves; ++i) s += red[i];
  __syncthreads();
  return s;
}

// out = relu( LN(agg + bias; g, be) + hres ) ; blockDim.x == C
__global__ __launch_bounds__(256) void ln_res_relu_kernel(
    const float* __restrict__ agg, const float* __restrict__ bias,
    const float* __restrict__ g, const float* __restrict__ be,
    const float* __restrict__ hres, float* __restrict__ out) {
  __shared__ float red[8];
  int C = blockDim.x;
  int nw = C >> 5;
  size_t idx = (size_t)blockIdx.x * C + threadIdx.x;
  float v = agg[idx] + bias[threadIdx.x];
  float mu = block_sum(v, red, nw) / (float)C;
  float dv = v - mu;
  float var = block_sum(dv * dv, red, nw) / (float)C;
  float y = dv * rsqrtf(var + 1e-5f) * g[threadIdx.x] + be[threadIdx.x];
  out[idx] = fmaxf(y + hres[idx], 0.f);
}

// x = LN(x + bias; g, be) in place ; blockDim.x == C
__global__ __launch_bounds__(256) void ln_inplace_kernel(
    float* __restrict__ x, const float* __restrict__ bias,
    const float* __restrict__ g, const float* __restrict__ be) {
  __shared__ float red[8];
  int C = blockDim.x;
  int nw = C >> 5;
  size_t idx = (size_t)blockIdx.x * C + threadIdx.x;
  float v = x[idx] + bias[threadIdx.x];
  float mu = block_sum(v, red, nw) / (float)C;
  float dv = v - mu;
  float var = block_sum(dv * dv, red, nw) / (float)C;
  x[idx] = dv * rsqrtf(var + 1e-5f) * g[threadIdx.x] + be[threadIdx.x];
}

// Row L2 normalize in place ; blockDim.x == C
__global__ __launch_bounds__(256) void l2norm_kernel(float* __restrict__ x) {
  __shared__ float red[8];
  int C = blockDim.x;
  int nw = C >> 5;
  size_t idx = (size_t)blockIdx.x * C + threadIdx.x;
  float v = x[idx];
  float ss = block_sum(v * v, red, nw);
  float nrm = fmaxf(sqrtf(ss), 1e-12f);
  x[idx] = v / nrm;
}

// ---------------------------------------------------------------------------
// Fills
// ---------------------------------------------------------------------------
__global__ __launch_bounds__(256) void fill_f32_kernel(float* p, long long n, float v) {
  long long i = (long long)blockIdx.x * blockDim.x + threadIdx.x;
  if (i < n) p[i] = v;
}
__global__ __launch_bounds__(256) void fill_u32_kernel(unsigned* p, long long n, unsigned v) {
  long long i = (long long)blockIdx.x * blockDim.x + threadIdx.x;
  if (i < n) p[i] = v;
}

// ---------------------------------------------------------------------------
// Host orchestration
// ---------------------------------------------------------------------------
extern "C" void kernel_launch(void* const* d_in, const int* in_sizes, int n_in,
                              void* d_out, int out_size, void* d_ws, size_t ws_size,
                              hipStream_t stream) {
  (void)n_in; (void)out_size; (void)ws_size;
  const float*     x   = (const float*)d_in[0];
  const long long* ei  = (const long long*)d_in[1];
  const float*     Wp  = (const float*)d_in[2];
  const float*     bp  = (const float*)d_in[3];
  const float*     W0  = (const float*)d_in[4];
  const float*     as0 = (const float*)d_in[5];
  const float*     ad0 = (const float*)d_in[6];
  const float*     b0  = (const float*)d_in[7];
  const float*     W1  = (const float*)d_in[8];
  const float*     as1 = (const float*)d_in[9];
  const float*     ad1 = (const float*)d_in[10];
  const float*     b1  = (const float*)d_in[11];
  const float*     g0  = (const float*)d_in[12];
  const float*     be0 = (const float*)d_in[13];
  const float*     gl1 = (const float*)d_in[14];
  const float*     be1 = (const float*)d_in[15];
  const float*     Wo  = (const float*)d_in[16];
  const float*     bo  = (const float*)d_in[17];

  const int INF = 256, HID = 256, OUTF = 128, H0 = 4, C0 = 64;
  const int Nn = in_sizes[0] / INF;
  const int E  = in_sizes[1] / 2;
  const long long Et = (long long)E + Nn;

  // ---- workspace carve (with reuse) ----
  char* ws = (char*)d_ws;
  size_t off = 0;
  auto carve = [&](size_t bytes) -> void* {
    void* p = ws + off;
    off += (bytes + 255) & ~(size_t)255;
    return p;
  };
  float*    bufA = (float*)carve((size_t)Nn * HID * 4);  // H, later AGG1/h2
  float*    bufB = (float*)carve((size_t)Nn * HID * 4);  // XW0, later H1
  float*    bufC = (float*)carve((size_t)Nn * HID * 4);  // AGG0, later XW1
  float*    ex   = (float*)carve((size_t)Et * H0 * 4);   // per-edge softmax numerators
  bf16_t*   WpF  = (bf16_t*)carve((size_t)INF * HID * 2);
  bf16_t*   W0F  = (bf16_t*)carve((size_t)HID * HID * 2);
  bf16_t*   W1F  = (bf16_t*)carve((size_t)HID * OUTF * 2);
  bf16_t*   WoF  = (bf16_t*)carve((size_t)OUTF * OUTF * 2);
  float*    als0 = (float*)carve((size_t)Nn * H0 * 4);
  float*    ald0 = (float*)carve((size_t)Nn * H0 * 4);
  unsigned* mx0  = (unsigned*)carve((size_t)Nn * H0 * 4);
  float*    dn0  = (float*)carve((size_t)Nn * H0 * 4);
  float*    als1 = (float*)carve((size_t)Nn * 4);
  float*    ald1 = (float*)carve((size_t)Nn * 4);
  unsigned* mx1  = (unsigned*)carve((size_t)Nn * 4);
  float*    dn1  = (float*)carve((size_t)Nn * 4);
  float*    outp = (float*)d_out;

  dim3 blk(256);
  const int gemmMB = (Nn + BM - 1) / BM;
  auto blocks1d = [](long long n, int t) { return (unsigned)((n + t - 1) / t); };
  const unsigned tb = blocks1d(Et, 256);      // thread-per-edge kernels
  const unsigned wb = blocks1d(Et, 8);        // wave-per-edge kernels (8 waves/blk)

  // ===== Pre-pack weights to bf16 fragment-major =====
  pack_weight_kernel<<<blocks1d((long long)INF * HID, 256), blk, 0, stream>>>(Wp, WpF, INF, HID);
  pack_weight_kernel<<<blocks1d((long long)HID * HID, 256), blk, 0, stream>>>(W0, W0F, HID, HID);
  pack_weight_kernel<<<blocks1d((long long)HID * OUTF, 256), blk, 0, stream>>>(W1, W1F, HID, OUTF);
  pack_weight_kernel<<<blocks1d((long long)OUTF * OUTF, 256), blk, 0, stream>>>(Wo, WoF, OUTF, OUTF);

  // ===== Projection: H = relu(x @ Wp + bp) =====
  wmma_gemm_kernel<2><<<dim3(gemmMB, HID / BN), blk, 0, stream>>>(
      x, WpF, bp, bufA, Nn, HID, INF);

  // ===== GAT layer 0 (4 heads x 64, concat) =====
  wmma_gemm_kernel<0><<<dim3(gemmMB, HID / BN), blk, 0, stream>>>(
      bufA, W0F, nullptr, bufB, Nn, HID, HID);                     // XW0
  attn_coef_kernel<<<blocks1d((long long)Nn * H0, 8), blk, 0, stream>>>(
      bufB, as0, ad0, als0, ald0, Nn, H0, C0);
  fill_u32_kernel<<<blocks1d((long long)Nn * H0, 256), blk, 0, stream>>>(
      mx0, (long long)Nn * H0, ORD_NEG_INF);
  fill_f32_kernel<<<blocks1d((long long)Nn * H0, 256), blk, 0, stream>>>(
      dn0, (long long)Nn * H0, 0.f);
  fill_f32_kernel<<<blocks1d((long long)Nn * HID, 256), blk, 0, stream>>>(
      bufC, (long long)Nn * HID, 0.f);
  edge_max_kernel<<<tb, blk, 0, stream>>>(ei, E, Nn, H0, als0, ald0, mx0);
  edge_expsum_kernel<<<tb, blk, 0, stream>>>(ei, E, Nn, H0, als0, ald0, mx0, ex, dn0);
  edge_agg_kernel<<<wb, blk, 0, stream>>>(ei, E, Nn, H0, C0, bufB, ex, dn0, bufC);
  // H1 = relu(LN(AGG0 + b0) + H)   (writes over XW0)
  ln_res_relu_kernel<<<Nn, HID, 0, stream>>>(bufC, b0, g0, be0, bufA, bufB);

  // ===== GAT layer 1 (1 head x 128, mean==identity) =====
  wmma_gemm_kernel<0><<<dim3(gemmMB, OUTF / BN), blk, 0, stream>>>(
      bufB, W1F, nullptr, bufC, Nn, OUTF, HID);                    // XW1
  attn_coef_kernel<<<blocks1d((long long)Nn, 8), blk, 0, stream>>>(
      bufC, as1, ad1, als1, ald1, Nn, 1, OUTF);
  fill_u32_kernel<<<blocks1d((long long)Nn, 256), blk, 0, stream>>>(
      mx1, (long long)Nn, ORD_NEG_INF);
  fill_f32_kernel<<<blocks1d((long long)Nn, 256), blk, 0, stream>>>(
      dn1, (long long)Nn, 0.f);
  fill_f32_kernel<<<blocks1d((long long)Nn * OUTF, 256), blk, 0, stream>>>(
      bufA, (long long)Nn * OUTF, 0.f);
  edge_max_kernel<<<tb, blk, 0, stream>>>(ei, E, Nn, 1, als1, ald1, mx1);
  edge_expsum_kernel<<<tb, blk, 0, stream>>>(ei, E, Nn, 1, als1, ald1, mx1, ex, dn1);
  edge_agg_kernel<<<wb, blk, 0, stream>>>(ei, E, Nn, 1, OUTF, bufC, ex, dn1, bufA);
  // h2 = LN(AGG1 + b1) in place
  ln_inplace_kernel<<<Nn, OUTF, 0, stream>>>(bufA, b1, gl1, be1);

  // ===== Output head: out = normalize(h2 @ Wo + bo) =====
  wmma_gemm_kernel<1><<<dim3(gemmMB, OUTF / BN), blk, 0, stream>>>(
      bufA, WoF, bo, outp, Nn, OUTF, OUTF);
  l2norm_kernel<<<Nn, OUTF, 0, stream>>>(outp);
}
